// GptOssExperts_71459665871173
// MI455X (gfx1250) — matrix operand order repaired and checked by reading, not compiled, not run on hardware
//
#include <hip/hip_runtime.h>
#include <hip/hip_bf16.h>

#define NE      8
#define HID     2880
#define IMD     2880
#define GUP     (2 * IMD)      // 5760
#define NTOK    2048
#define ALPHA   1.702f
#define FLIMIT  7.0f

#define BM 128
#define BN 128
#define BK 32
#define LDAB 40    // bf16 pitch, A tile rows (32 + 8 pad, 80B)
#define LDBB 136   // bf16 pitch, B tile rows (128 + 8 pad, 272B)

typedef __bf16  v4bf  __attribute__((ext_vector_type(4)));
typedef __bf16  v8bf  __attribute__((ext_vector_type(8)));
typedef __bf16  v16bf __attribute__((ext_vector_type(16)));
typedef float   v8f   __attribute__((ext_vector_type(8)));
typedef int     v4i   __attribute__((ext_vector_type(4)));

typedef __attribute__((address_space(1))) v4i* gas_v4i_ptr;  // global (AS1)
typedef __attribute__((address_space(3))) v4i* las_v4i_ptr;  // LDS (AS3)

#if defined(__has_builtin)
#if __has_builtin(__builtin_amdgcn_global_load_async_to_lds_b128)
#define ASYNC_A 1
#endif
#endif
#ifndef ASYNC_A
#define ASYNC_A 0
#endif

__device__ __forceinline__ v4bf cvt4(float4 v) {
    v4bf t;
    t[0] = (__bf16)v.x; t[1] = (__bf16)v.y; t[2] = (__bf16)v.z; t[3] = (__bf16)v.w;
    return t;
}
__device__ __forceinline__ v16bf frag16(const __bf16* lo, const __bf16* hi) {
    v8bf a = *(const v8bf*)lo;
    v8bf b = *(const v8bf*)hi;
    return __builtin_shufflevector(a, b, 0,1,2,3,4,5,6,7,8,9,10,11,12,13,14,15);
}
__device__ __forceinline__ void async_copy16(const __bf16* gp, __bf16* lp) {
#if ASYNC_A
    __builtin_amdgcn_global_load_async_to_lds_b128(
        (gas_v4i_ptr)gp, (las_v4i_ptr)lp, 0, 0);
#endif
}
__device__ __forceinline__ void wait_async0() {
#if ASYNC_A
#if __has_builtin(__builtin_amdgcn_s_wait_asynccnt)
    __builtin_amdgcn_s_wait_asynccnt(0);
#else
    asm volatile("s_wait_asynccnt 0x0" ::: "memory");
#endif
#endif
}

// ---------------------------------------------------------------------------
// f32 -> bf16 pre-conversion of hidden_states (one-shot, ~1.5us of HBM)
// ---------------------------------------------------------------------------
__global__ __launch_bounds__(256)
void cvt_bf16_kernel(const float* __restrict__ src, __bf16* __restrict__ dst, int n)
{
    const int i = (blockIdx.x * 256 + threadIdx.x) * 8;
    if (i < n) {
        float4 a = *(const float4*)(src + i);
        float4 b = *(const float4*)(src + i + 4);
        v4bf t0 = cvt4(a), t1 = cvt4(b);
        *(v8bf*)(dst + i) = __builtin_shufflevector(t0, t1, 0,1,2,3,4,5,6,7);
    }
}

// A tile 128x32 bf16 = 512 x 16B chunks; 4 chunks/thread (128 threads)
#define A_CHUNKS 4

// ---------------------------------------------------------------------------
// GEMM1 + fused clamped-GLU.  gate = col 0::2, up = col 1::2.  G stored bf16.
// ---------------------------------------------------------------------------
__global__ __launch_bounds__(128)
void gemm1_act_kernel(const __bf16* __restrict__ Xb,   // [NTOK, HID] bf16
                      const float*  __restrict__ W,    // [NE, HID, GUP]
                      const float*  __restrict__ BIAS, // [NE, GUP]
                      __bf16* __restrict__ Gb,         // [NTOK, IMD] bf16 (ws)
                      int e)
{
    __shared__ __bf16 As[2][BM * LDAB];
    __shared__ __bf16 Bs[2][BK * LDBB];

    const int tid  = threadIdx.x;
    const int lane = tid & 31;
    const int wave = tid >> 5;
    const int wm   = wave & 1;          // 2x2 waves, 64x64 tile each
    const int wn   = wave >> 1;
    const int lm   = lane & 15;
    const int lh   = lane >> 4;

    const int m0 = blockIdx.y * BM;
    const int n0 = blockIdx.x * BN;

    const float* Wg = W + (size_t)e * HID * GUP;

#if !ASYNC_A
    uint4 pa[A_CHUNKS];
#endif
    float4 pb[8];

    // ---- A tile: raw bf16 copy global -> LDS (async path: DMA, no VGPR stage)
    auto stageA = [&](int k0, int buf) {
#pragma unroll
        for (int i = 0; i < A_CHUNKS; ++i) {
            const int idx = tid + 128 * i;
            const int r = idx >> 2, c = (idx & 3) << 3;      // c: bf16 elems
            const __bf16* gp = Xb + (size_t)(m0 + r) * HID + k0 + c;
            __bf16* lp = &As[buf][r * LDAB + c];
#if ASYNC_A
            async_copy16(gp, lp);
#else
            pa[i] = *(const uint4*)gp;
            (void)lp;
#endif
        }
    };
#if !ASYNC_A
    auto commitA = [&](int buf) {
#pragma unroll
        for (int i = 0; i < A_CHUNKS; ++i) {
            const int idx = tid + 128 * i;
            const int r = idx >> 2, c = (idx & 3) << 3;
            *(uint4*)(&As[buf][r * LDAB + c]) = pa[i];
        }
    };
#endif
    // ---- B tile: f32 global -> regs, cvt once -> bf16 LDS (b128 stores)
    auto stageB = [&](int k0) {
#pragma unroll
        for (int i = 0; i < 4; ++i) {
            const int idx = tid + 128 * i;
            const int r = idx >> 4, c8 = (idx & 15) << 3;    // 8-float chunk
            const float* gp = Wg + (size_t)(k0 + r) * GUP + n0 + c8;
            pb[2*i]   = *(const float4*)(gp);
            pb[2*i+1] = *(const float4*)(gp + 4);
        }
    };
    auto commitB = [&](int buf) {
#pragma unroll
        for (int i = 0; i < 4; ++i) {
            const int idx = tid + 128 * i;
            const int r = idx >> 4, c8 = (idx & 15) << 3;
            v4bf t0 = cvt4(pb[2*i]), t1 = cvt4(pb[2*i+1]);
            *(v8bf*)(&Bs[buf][r * LDBB + c8]) =
                __builtin_shufflevector(t0, t1, 0,1,2,3,4,5,6,7);
        }
    };

    v8f acc[4][4] = {};

    stageA(0, 0);
    stageB(0);
#if !ASYNC_A
    commitA(0);
#endif
    commitB(0);
    wait_async0();
    __syncthreads();

    int cur = 0;
    for (int k0 = 0; k0 < HID; k0 += BK) {
        const bool nxt = (k0 + BK) < HID;
        if (nxt) {
            stageB(k0 + BK);            // f32 loads overlap compute
            stageA(k0 + BK, cur ^ 1);   // async: DMA straight into other buffer
        }

        auto loadA = [&](int mt) -> v16bf {
            const __bf16* ap = &As[cur][(wm * 64 + mt * 16 + lm) * LDAB + lh * 8];
            return frag16(ap, ap + 16);
        };

        v16bf bfrag[4];
#pragma unroll
        for (int nt = 0; nt < 4; ++nt) {
            const __bf16* bp = &Bs[cur][(lh * 8) * LDBB + wn * 64 + nt * 16 + lm];
#pragma unroll
            for (int s = 0; s < 16; ++s)
                bfrag[nt][s] = bp[(s < 8 ? s : s + 8) * LDBB];
        }

        v16bf af = loadA(0);            // software-pipelined A fragment
#pragma unroll
        for (int mt = 0; mt < 4; ++mt) {
            v16bf afn = af;
            if (mt < 3) afn = loadA(mt + 1);   // issue before this group's WMMAs
#pragma unroll
            for (int nt = 0; nt < 4; ++nt)
                acc[mt][nt] = __builtin_amdgcn_wmma_f32_16x16x32_bf16(
                    false, af, false, bfrag[nt], (short)0, acc[mt][nt], false, false);
            af = afn;
        }

        if (nxt) {
#if !ASYNC_A
            commitA(cur ^ 1);
#endif
            commitB(cur ^ 1);
            wait_async0();
        }
        __syncthreads();
        cur ^= 1;
    }

    const float* bias = BIAS + (size_t)e * GUP;
#pragma unroll
    for (int mt = 0; mt < 4; ++mt) {
#pragma unroll
        for (int nt = 0; nt < 4; ++nt) {
            const int col = n0 + wn * 64 + nt * 16 + lm;
            const float bc = bias[col];
#pragma unroll
            for (int r = 0; r < 8; ++r) {
                const int row = m0 + wm * 64 + mt * 16 + (lh ? r + 8 : r);
                const float c = acc[mt][nt][r] + bc;
                const float partner = __shfl_xor(c, 1, 32);
                if ((lane & 1) == 0) {
                    float gate = fminf(c, FLIMIT);
                    float up   = fminf(fmaxf(partner, -FLIMIT), FLIMIT);
                    float glu  = gate / (1.0f + __expf(-ALPHA * gate));
                    Gb[(size_t)row * IMD + (col >> 1)] = (__bf16)((up + 1.0f) * glu);
                }
            }
        }
    }
}

// ---------------------------------------------------------------------------
// GEMM2 + routing-weighted accumulate (experts serialized on the stream)
// ---------------------------------------------------------------------------
__global__ __launch_bounds__(128)
void gemm2_acc_kernel(const __bf16* __restrict__ Gb,   // [NTOK, IMD] bf16
                      const float*  __restrict__ W,    // [NE, IMD, HID]
                      const float*  __restrict__ BIAS, // [NE, HID]
                      const float*  __restrict__ RW,   // [NTOK, NE]
                      float* __restrict__ OUT,         // [NTOK, HID]
                      int e)
{
    __shared__ __bf16 As[2][BM * LDAB];
    __shared__ __bf16 Bs[2][BK * LDBB];

    const int tid  = threadIdx.x;
    const int lane = tid & 31;
    const int wave = tid >> 5;
    const int wm   = wave & 1;
    const int wn   = wave >> 1;
    const int lm   = lane & 15;
    const int lh   = lane >> 4;

    const int m0 = blockIdx.y * BM;
    const int n0 = blockIdx.x * BN;     // last tile partial (2880 % 128 != 0)

    const float* Wd = W + (size_t)e * IMD * HID;

#if !ASYNC_A
    uint4 pa[A_CHUNKS];
#endif
    float4 pb[8];

    auto stageA = [&](int k0, int buf) {
#pragma unroll
        for (int i = 0; i < A_CHUNKS; ++i) {
            const int idx = tid + 128 * i;
            const int r = idx >> 2, c = (idx & 3) << 3;
            const __bf16* gp = Gb + (size_t)(m0 + r) * IMD + k0 + c;
            __bf16* lp = &As[buf][r * LDAB + c];
#if ASYNC_A
            async_copy16(gp, lp);
#else
            pa[i] = *(const uint4*)gp;
            (void)lp;
#endif
        }
    };
#if !ASYNC_A
    auto commitA = [&](int buf) {
#pragma unroll
        for (int i = 0; i < A_CHUNKS; ++i) {
            const int idx = tid + 128 * i;
            const int r = idx >> 2, c = (idx & 3) << 3;
            *(uint4*)(&As[buf][r * LDAB + c]) = pa[i];
        }
    };
#endif
    auto stageB = [&](int k0) {
#pragma unroll
        for (int i = 0; i < 4; ++i) {
            const int idx = tid + 128 * i;
            const int r = idx >> 4, c8 = (idx & 15) << 3;
            pb[2*i]   = make_float4(0.f, 0.f, 0.f, 0.f);
            pb[2*i+1] = make_float4(0.f, 0.f, 0.f, 0.f);
            if (n0 + c8 < HID) {        // HID % 8 == 0: chunk fully in or out
                const float* gp = Wd + (size_t)(k0 + r) * HID + n0 + c8;
                pb[2*i]   = *(const float4*)(gp);
                pb[2*i+1] = *(const float4*)(gp + 4);
            }
        }
    };
    auto commitB = [&](int buf) {
#pragma unroll
        for (int i = 0; i < 4; ++i) {
            const int idx = tid + 128 * i;
            const int r = idx >> 4, c8 = (idx & 15) << 3;
            v4bf t0 = cvt4(pb[2*i]), t1 = cvt4(pb[2*i+1]);
            *(v8bf*)(&Bs[buf][r * LDBB + c8]) =
                __builtin_shufflevector(t0, t1, 0,1,2,3,4,5,6,7);
        }
    };

    v8f acc[4][4] = {};

    stageA(0, 0);
    stageB(0);
#if !ASYNC_A
    commitA(0);
#endif
    commitB(0);
    wait_async0();
    __syncthreads();

    int cur = 0;
    for (int k0 = 0; k0 < IMD; k0 += BK) {
        const bool nxt = (k0 + BK) < IMD;
        if (nxt) {
            stageB(k0 + BK);
            stageA(k0 + BK, cur ^ 1);
        }

        auto loadA = [&](int mt) -> v16bf {
            const __bf16* ap = &As[cur][(wm * 64 + mt * 16 + lm) * LDAB + lh * 8];
            return frag16(ap, ap + 16);
        };

        v16bf bfrag[4];
#pragma unroll
        for (int nt = 0; nt < 4; ++nt) {
            const __bf16* bp = &Bs[cur][(lh * 8) * LDBB + wn * 64 + nt * 16 + lm];
#pragma unroll
            for (int s = 0; s < 16; ++s)
                bfrag[nt][s] = bp[(s < 8 ? s : s + 8) * LDBB];
        }

        v16bf af = loadA(0);
#pragma unroll
        for (int mt = 0; mt < 4; ++mt) {
            v16bf afn = af;
            if (mt < 3) afn = loadA(mt + 1);
#pragma unroll
            for (int nt = 0; nt < 4; ++nt)
                acc[mt][nt] = __builtin_amdgcn_wmma_f32_16x16x32_bf16(
                    false, af, false, bfrag[nt], (short)0, acc[mt][nt], false, false);
            af = afn;
        }

        if (nxt) {
#if !ASYNC_A
            commitA(cur ^ 1);
#endif
            commitB(cur ^ 1);
            wait_async0();
        }
        __syncthreads();
        cur ^= 1;
    }

    const float* bias = BIAS + (size_t)e * HID;
#pragma unroll
    for (int mt = 0; mt < 4; ++mt) {
#pragma unroll
        for (int r = 0; r < 8; ++r) {
            const int row = m0 + wm * 64 + mt * 16 + (lh ? r + 8 : r);
            const float rw = RW[(size_t)row * NE + e];
#pragma unroll
            for (int nt = 0; nt < 4; ++nt) {
                const int col = n0 + wn * 64 + nt * 16 + lm;
                if (col < HID)
                    OUT[(size_t)row * HID + col] += rw * (acc[mt][nt][r] + bias[col]);
            }
        }
    }
}

// ---------------------------------------------------------------------------
extern "C" void kernel_launch(void* const* d_in, const int* in_sizes, int n_in,
                              void* d_out, int out_size, void* d_ws, size_t ws_size,
                              hipStream_t stream)
{
    const float* X   = (const float*)d_in[0];  // hidden_states  [NTOK, HID]
    const float* RW  = (const float*)d_in[1];  // routing_weights[NTOK, NE]
    const float* Wgu = (const float*)d_in[2];  // gate_up_proj   [NE, HID, GUP]
    const float* Bgu = (const float*)d_in[3];  // gate_up_bias   [NE, GUP]
    const float* Wd  = (const float*)d_in[4];  // down_proj      [NE, IMD, HID]
    const float* Bd  = (const float*)d_in[5];  // down_bias      [NE, HID]
    float* OUT = (float*)d_out;                // [NTOK, HID] f32

    __bf16* Xb = (__bf16*)d_ws;                                    // 11.8 MB
    __bf16* Gb = (__bf16*)((char*)d_ws + (size_t)NTOK * HID * 2);  // 11.8 MB

    (void)in_sizes; (void)n_in; (void)ws_size;

    (void)hipMemsetAsync(d_out, 0, (size_t)out_size * sizeof(float), stream);

    const int nx = NTOK * HID;
    cvt_bf16_kernel<<<nx / (256 * 8), 256, 0, stream>>>(X, Xb, nx);

    dim3 block(128);
    dim3 g1(GUP / BN, NTOK / BM);              // 45 x 16
    dim3 g2((HID + BN - 1) / BN, NTOK / BM);   // 23 x 16

    for (int e = 0; e < NE; ++e) {
        gemm1_act_kernel<<<g1, block, 0, stream>>>(Xb, Wgu, Bgu, Gb, e);
        gemm2_acc_kernel<<<g2, block, 0, stream>>>(Gb, Wd, Bd, RW, OUT, e);
    }
}
